// Sampler_69277822484525
// MI455X (gfx1250) — matrix operand order repaired and checked by reading, not compiled, not run on hardware
//
#include <hip/hip_runtime.h>
#include <hip/hip_bf16.h>
#include <stdint.h>

#define IN_DIM  512
#define LATENT  128
#define EMBED   10000
#define KSTEPS  (IN_DIM / 32)          // 16 K-chunks of 32
#define CHUNK_BYTES (2 * 2 * 8 * 32 * 32)  // heads*hilo*mtiles*lanes*32B = 32 KB

typedef __bf16 bf16_t;
typedef __bf16 v16bf __attribute__((ext_vector_type(16)));
typedef float  v8f   __attribute__((ext_vector_type(8)));
typedef int    v4i   __attribute__((vector_size(16)));   // match builtin's pointee type

#if __has_builtin(__builtin_amdgcn_global_load_async_to_lds_b128)
#  define HAVE_ASYNC_LDS 1
#else
#  define HAVE_ASYNC_LDS 0
#endif

typedef v4i __attribute__((address_space(1)))* gptr_v4i;  // global (device) ptr
typedef v4i __attribute__((address_space(3)))* lptr_v4i;  // LDS ptr

// ---------------------------------------------------------------------------
// Kernel 1: pack W_logvar / W_mean (f32 [128][512]) into bf16 hi/lo fragments,
// already in the WMMA 16-bit A-matrix lane layout (16x32 tile per fragment):
//   lanes 0-15 : row = lane,   K = {0..7, 16..23}
//   lanes 16-31: row = lane-16,K = {8..15, 24..31}
// ws linear index: tid = kstep*1024 + head*512 + hilo*256 + mtile*32 + lane,
// each tid owns 32 bytes (16 bf16).
// ---------------------------------------------------------------------------
__global__ __launch_bounds__(256) void pack_weights(
    const float* __restrict__ Wlv, const float* __restrict__ Wmu,
    bf16_t* __restrict__ wsfrag) {
  int tid   = blockIdx.x * blockDim.x + threadIdx.x;  // 0..16383
  int lane  = tid & 31;
  int mtile = (tid >> 5) & 7;
  int hilo  = (tid >> 8) & 1;
  int head  = (tid >> 9) & 1;
  int kstep = tid >> 10;
  const float* W = head ? Wmu : Wlv;
  int row   = mtile * 16 + (lane & 15);
  int kbase = kstep * 32 + ((lane >> 4) << 3);  // +0 or +8
  bf16_t v[16];
#pragma unroll
  for (int j = 0; j < 8; ++j) {
    float f0 = W[row * IN_DIM + kbase + j];
    float f1 = W[row * IN_DIM + kbase + 16 + j];
    bf16_t h0 = (bf16_t)f0, h1 = (bf16_t)f1;
    bf16_t l0 = (bf16_t)(f0 - (float)h0);   // exact residual, then bf16 round
    bf16_t l1 = (bf16_t)(f1 - (float)h1);
    v[j]     = hilo ? l0 : h0;
    v[8 + j] = hilo ? l1 : h1;
  }
  uint4* dst = (uint4*)(wsfrag + (size_t)tid * 16);
  const uint4* s = (const uint4*)v;
  dst[0] = s[0];
  dst[1] = s[1];
}

// ---------------------------------------------------------------------------
// Kernel 2: fused dual-head GEMM. Block = 256 threads (8 waves), covers
// 128 batch rows x 128 latents x 2 heads. Wave w owns batch rows
// [block*128 + 16w, +16): B-matrix = x (bf16 hi/lo built in-register),
// A-matrix = prepacked weights staged per-K-chunk into LDS.
// bf16x3: acc += Ahi*Bhi + Ahi*Blo + Alo*Bhi   (f32 accumulate)
// ---------------------------------------------------------------------------
__global__ __launch_bounds__(256) void gemm_heads(
    const float* __restrict__ x, const bf16_t* __restrict__ wsfrag,
    const float* __restrict__ b_lv, const float* __restrict__ b_mu,
    float* __restrict__ out0, float* __restrict__ out1) {
  __shared__ alignas(32) char lds[CHUNK_BYTES];

  const int t     = threadIdx.x;
  const int wave  = t >> 5;
  const int lane  = t & 31;
  const int lhalf = lane >> 4;  // 0/1 -> K half / latent sub-block
  const int lmod  = lane & 15;  // batch column within tile
  const long b0   = (long)blockIdx.x * 128 + wave * 16;

  v8f acc[16];
#pragma unroll
  for (int i = 0; i < 16; ++i) acc[i] = (v8f){0.f, 0.f, 0.f, 0.f, 0.f, 0.f, 0.f, 0.f};

  // lane's x stream: row b0+lmod, K = {lhalf*8..+7, 16+lhalf*8..+7}
  const float* xrow = x + (b0 + lmod) * IN_DIM + lhalf * 8;

  for (int ks = 0; ks < KSTEPS; ++ks) {
    __syncthreads();  // previous chunk fully consumed before overwrite

    // ---- stage 32 KB weight-fragment chunk: global -> LDS ----
    const char* gsrc = (const char*)wsfrag + (size_t)ks * CHUNK_BYTES;
#if HAVE_ASYNC_LDS
    {
      gptr_v4i gp = (gptr_v4i)gsrc;
      lptr_v4i lp = (lptr_v4i)lds;
#pragma unroll
      for (int i = 0; i < 8; ++i) {
        int idx = i * 256 + t;   // 16-byte units
        __builtin_amdgcn_global_load_async_to_lds_b128(gp + idx, lp + idx, 0, 0);
      }
    }
#if __has_builtin(__builtin_amdgcn_s_wait_asynccnt)
    __builtin_amdgcn_s_wait_asynccnt(0);
#else
    asm volatile("s_wait_asynccnt 0" ::: "memory");
#endif
#else
    {
      const uint4* gs = (const uint4*)gsrc;
      uint4* ld = (uint4*)lds;
#pragma unroll
      for (int i = 0; i < 8; ++i) ld[i * 256 + t] = gs[i * 256 + t];
    }
#endif
    __syncthreads();

    // ---- build x B-fragments (16x16 batch tile, K=32) as bf16 hi/lo ----
    const float4* xp = (const float4*)(xrow + ks * 32);
    float4 p0 = xp[0], p1 = xp[1], p2 = xp[4], p3 = xp[5];
    if (ks + 1 < KSTEPS) __builtin_prefetch(xrow + (ks + 1) * 32, 0, 0);
    float xs[16] = {p0.x, p0.y, p0.z, p0.w, p1.x, p1.y, p1.z, p1.w,
                    p2.x, p2.y, p2.z, p2.w, p3.x, p3.y, p3.z, p3.w};
    v16bf bh, bl;
#pragma unroll
    for (int e = 0; e < 16; ++e) {
      bf16_t h = (bf16_t)xs[e];
      bh[e] = h;
      bl[e] = (bf16_t)(xs[e] - (float)h);
    }

    // ---- 16 output tiles x 3 WMMAs each ----
    const char* lbase = lds + lane * 32;
#pragma unroll
    for (int head = 0; head < 2; ++head) {
#pragma unroll
      for (int m = 0; m < 8; ++m) {
        v16bf ahi = *(const v16bf*)(lbase + (size_t)((head * 2 + 0) * 8 + m) * 1024);
        v16bf alo = *(const v16bf*)(lbase + (size_t)((head * 2 + 1) * 8 + m) * 1024);
        v8f c = acc[head * 8 + m];
        c = __builtin_amdgcn_wmma_f32_16x16x32_bf16(false, ahi, false, bh, (short)0, c, false, false);
        c = __builtin_amdgcn_wmma_f32_16x16x32_bf16(false, ahi, false, bl, (short)0, c, false, false);
        c = __builtin_amdgcn_wmma_f32_16x16x32_bf16(false, alo, false, bh, (short)0, c, false, false);
        acc[head * 8 + m] = c;
      }
    }
  }

  // ---- epilogue: bias add + coalesced stores ----
  // C layout: lane (0-15 | 16-31) = batch col lmod, VGPR v = latent m*16 + lhalf*8 + v
#pragma unroll
  for (int head = 0; head < 2; ++head) {
    const float* bias = head ? b_mu : b_lv;
    float* outp = head ? out1 : out0;
#pragma unroll
    for (int m = 0; m < 8; ++m) {
      int latbase = m * 16 + lhalf * 8;
      float4 bb0 = *(const float4*)(bias + latbase);
      float4 bb1 = *(const float4*)(bias + latbase + 4);
      v8f c = acc[head * 8 + m];
      float4 r0 = {c[0] + bb0.x, c[1] + bb0.y, c[2] + bb0.z, c[3] + bb0.w};
      float4 r1 = {c[4] + bb1.x, c[5] + bb1.y, c[6] + bb1.z, c[7] + bb1.w};
      float* o = outp + (b0 + lmod) * LATENT + latbase;
      *(float4*)o = r0;
      *(float4*)(o + 4) = r1;
    }
  }
}

// ---------------------------------------------------------------------------
// Kernel 3: embed[b][l] = W_embed[l][y[b]] + b_embed[l]
// Write-coalesced; 5 MB table stays in L2; y[b] is wave-uniform.
// ---------------------------------------------------------------------------
__global__ __launch_bounds__(256) void embed_gather(
    const int* __restrict__ y, const float* __restrict__ We,
    const float* __restrict__ be, float* __restrict__ out2, int total) {
  int tid = blockIdx.x * blockDim.x + threadIdx.x;
  if (tid >= total) return;
  int b = tid >> 7;
  int l = tid & 127;
  int idx = y[b];
  out2[tid] = We[l * EMBED + idx] + be[l];
}

extern "C" void kernel_launch(void* const* d_in, const int* in_sizes, int n_in,
                              void* d_out, int out_size, void* d_ws, size_t ws_size,
                              hipStream_t stream) {
  (void)n_in; (void)out_size; (void)ws_size;
  const float* x   = (const float*)d_in[0];
  const int*   y   = (const int*)d_in[1];
  const float* Wlv = (const float*)d_in[2];
  const float* blv = (const float*)d_in[3];
  const float* Wmu = (const float*)d_in[4];
  const float* bmu = (const float*)d_in[5];
  const float* We  = (const float*)d_in[6];
  const float* be  = (const float*)d_in[7];

  const int Bn = in_sizes[1];  // 262144
  float* out0 = (float*)d_out;
  float* out1 = out0 + (size_t)Bn * LATENT;
  float* out2 = out0 + 2 * (size_t)Bn * LATENT;
  bf16_t* wsfrag = (bf16_t*)d_ws;  // 512 KB of packed weight fragments

  pack_weights<<<64, 256, 0, stream>>>(Wlv, Wmu, wsfrag);
  gemm_heads<<<Bn / 128, 256, 0, stream>>>(x, wsfrag, blv, bmu, out0, out1);
  int total = Bn * LATENT;
  embed_gather<<<(total + 255) / 256, 256, 0, stream>>>(y, We, be, out2, total);
}
